// DriftingPolicy_85710367359179
// MI455X (gfx1250) — matrix operand order repaired and checked by reading, not compiled, not run on hardware
//
#include <hip/hip_runtime.h>
#include <hip/hip_bf16.h>

// MI455X / gfx1250 fused contrastive-drift loss.
// dist_pos/dist_neg (128MB, ~fits 192MB L2) materialized once via WMMA f16
// GEMM; the 3 temperatures fold algebraically into one effective matrix per
// side; aggregation = 2 WMMA GEMMs with A fragments built cooperatively in
// LDS (exp/normalize computed once per element per block, not per wave).

static constexpr int  kN  = 4096;
static constexpr int  kD  = 256;
static constexpr int  kT  = 3;

typedef _Float16 half_t;
typedef __attribute__((ext_vector_type(16))) _Float16 v16h;
typedef __attribute__((ext_vector_type(8)))  _Float16 v8h;
typedef __attribute__((ext_vector_type(2)))  _Float16 v2h;
typedef __attribute__((ext_vector_type(8)))  float    v8f;

__device__ __forceinline__ float wave_sum(float v) {
  #pragma unroll
  for (int o = 16; o > 0; o >>= 1) v += __shfl_xor(v, o, 32);
  return v;
}

// ---------------------------------------------------------------- zero init
__global__ void k_zero(float* p, int n, float* out) {
  int i = blockIdx.x * blockDim.x + threadIdx.x;
  if (i == 0) out[0] = 0.f;
  for (; i < n; i += gridDim.x * blockDim.x) p[i] = 0.f;
}

// ------------------------------------------- fp16 staging + norms + y^T
__global__ void k_prep(const float* __restrict__ x, const float* __restrict__ yp,
                       const float* __restrict__ yn,
                       half_t* xh, half_t* yph, half_t* ynh,
                       half_t* ypT, half_t* ynT,
                       float* nx, float* nyp, float* nyn) {
  int b = blockIdx.x;
  int which = b >> 12;            // 0=x, 1=y_pos, 2=y_neg   (4096 rows each)
  int row   = b & (kN - 1);
  int t = threadIdx.x;            // 256 threads = one row
  const float* src = (which == 0) ? x : (which == 1) ? yp : yn;
  half_t*      dst = (which == 0) ? xh : (which == 1) ? yph : ynh;
  float v = src[(size_t)row * kD + t];
  half_t h = (half_t)v;
  dst[(size_t)row * kD + t] = h;
  if (which == 1) ypT[(size_t)t * kN + row] = h;
  if (which == 2) ynT[(size_t)t * kN + row] = h;
  float s = wave_sum(v * v);
  __shared__ float red[8];
  int wave = t >> 5, lane = t & 31;
  if (lane == 0) red[wave] = s;
  __syncthreads();
  if (t == 0) {
    float tot = 0.f;
    #pragma unroll
    for (int i = 0; i < 8; i++) tot += red[i];
    ((which == 0) ? nx : (which == 1) ? nyp : nyn)[row] = tot;
  }
}

// -------------------------------- dist GEMM: G = x@y^T, dist = sqrt(|x|^2+|y|^2-2G)
// grid (kN/128, kN/16, 2); block 256 = 8 waves; wave -> 16x16 tile of G.
__global__ void k_dist(const half_t* __restrict__ xh, const half_t* __restrict__ yph,
                       const half_t* __restrict__ ynh,
                       const float* __restrict__ nx, const float* __restrict__ nyp,
                       const float* __restrict__ nyn,
                       float* distP, float* distN, float* meansum) {
  int tid = threadIdx.x, wave = tid >> 5, lane = tid & 31;
  int i0 = blockIdx.y * 16;
  int j0 = blockIdx.x * 128 + wave * 16;
  bool pos = (blockIdx.z == 1);
  const half_t* yh  = pos ? yph : ynh;
  const float*  ny  = pos ? nyp : nyn;
  float*        dst = pos ? distP : distN;

  int am   = lane & 15;
  int asel = (lane & 16) ? 16 : 0;    // A: hi lanes take K+16
  int bsel = (lane & 16) ? 16 : 0;    // B: hi lanes take K+16
  const half_t* arow = xh + (size_t)(i0 + am) * kD;
  const half_t* brow = yh + (size_t)(j0 + am) * kD;

  v8f acc = {};
  for (int d0 = 0; d0 < kD; d0 += 32) {
    union { v16h v; v8h h[2]; } A;
    A.h[0] = *(const v8h*)(arow + d0 + asel);        // K 0..7 (+16 hi)
    A.h[1] = *(const v8h*)(arow + d0 + 8 + asel);    // K 8..15 (+16 hi)
    v16h B = *(const v16h*)(brow + d0 + bsel);       // K 0..15 / 16..31
    acc = __builtin_amdgcn_wmma_f32_16x16x32_f16(false, A.v, false, B,
                                                 (short)0, acc, false, false);
  }
  int n = j0 + am;
  float nyv = ny[n];
  int mbase = i0 + ((lane & 16) ? 8 : 0);
  float psum = 0.f;
  #pragma unroll
  for (int r = 0; r < 8; r++) {
    int m = mbase + r;
    float d2 = nx[m] + nyv - 2.f * acc[r];
    float dv = sqrtf(fmaxf(d2, 0.f));
    if (!pos && m == n) dv += 1000000.f;   // MASK_SELF on dist_neg diagonal
    dst[(size_t)m * kN + n] = dv;
    psum += dv;
  }
  if (pos) {                                // accumulate mean(dist_pos)
    float s = wave_sum(psum);
    if (lane == 0) atomicAdd(meansum, s);
  }
}

// ------------------------- row sums (over 8192 concat cols) + col sums, 3 temps
// grid (kN/256, kN/64, 2); block 256; tile = 64 rows x 256 cols.
__global__ void k_sums(const float* __restrict__ distP, const float* __restrict__ distN,
                       const float* __restrict__ meansum,
                       float* rs, float* csn, float* csp) {
  const float Ts[kT] = {0.05f, 0.1f, 0.2f};
  int tid = threadIdx.x, lane = tid & 31;
  bool pos = (blockIdx.z == 1);
  const float* dist = pos ? distP : distN;
  float* cs = pos ? csp : csn;
  float mean = meansum[0] * (1.f / ((float)kN * (float)kN));
  float ninv[kT];
  #pragma unroll
  for (int t = 0; t < kT; t++) ninv[t] = -1.f / (Ts[t] * mean);

  int c  = blockIdx.x * 256 + tid;
  int r0 = blockIdx.y * 64;
  __shared__ float srow[kT * 64];
  for (int i = tid; i < kT * 64; i += 256) srow[i] = 0.f;
  __syncthreads();

  float csl[kT] = {0.f, 0.f, 0.f};
  for (int r = 0; r < 64; r++) {
    float dv = dist[(size_t)(r0 + r) * kN + c];
    #pragma unroll
    for (int t = 0; t < kT; t++) {
      float e = __expf(fmaxf(dv * ninv[t], -80.f));
      csl[t] += e;
      float s = wave_sum(e);
      if (lane == 0) atomicAdd(&srow[t * 64 + r], s);
    }
  }
  __syncthreads();
  for (int i = tid; i < kT * 64; i += 256) {
    int t = i >> 6, r = i & 63;
    atomicAdd(&rs[(size_t)t * kN + r0 + r], srow[i]);
  }
  #pragma unroll
  for (int t = 0; t < kT; t++) atomicAdd(&cs[(size_t)t * kN + c], csl[t]);
}

// rs|csn|csp -> wrs|wcsn|wcsp  (contiguous blocks)
__global__ void k_rsqrt(const float* __restrict__ in, float* out, int n) {
  int i = blockIdx.x * blockDim.x + threadIdx.x;
  if (i < n) out[i] = rsqrtf(in[i]);
}

// ---------- weighted row sums: Sneg[t,i]=sum_j min(wrs*wcsn,1e6)*k_neg, Spos likewise
__global__ void k_wsums(const float* __restrict__ distP, const float* __restrict__ distN,
                        const float* __restrict__ meansum, const float* __restrict__ wrs,
                        const float* __restrict__ wcsn, const float* __restrict__ wcsp,
                        float* Sneg, float* Spos) {
  const float Ts[kT] = {0.05f, 0.1f, 0.2f};
  int tid = threadIdx.x, lane = tid & 31;
  bool pos = (blockIdx.z == 1);
  const float* dist = pos ? distP : distN;
  const float* wcs  = pos ? wcsp : wcsn;
  float* S          = pos ? Spos : Sneg;
  float mean = meansum[0] * (1.f / ((float)kN * (float)kN));
  float ninv[kT];
  #pragma unroll
  for (int t = 0; t < kT; t++) ninv[t] = -1.f / (Ts[t] * mean);

  int c  = blockIdx.x * 256 + tid;
  int r0 = blockIdx.y * 64;
  float wc[kT];
  #pragma unroll
  for (int t = 0; t < kT; t++) wc[t] = wcs[(size_t)t * kN + c];

  __shared__ float srow[kT * 64];
  for (int i = tid; i < kT * 64; i += 256) srow[i] = 0.f;
  __syncthreads();

  for (int r = 0; r < 64; r++) {
    float dv = dist[(size_t)(r0 + r) * kN + c];
    #pragma unroll
    for (int t = 0; t < kT; t++) {
      float e = __expf(fmaxf(dv * ninv[t], -80.f));
      float wr = wrs[(size_t)t * kN + r0 + r];                  // scalar bcast
      float term = fminf(wr * wc[t], 1e6f) * e;                 // == rsqrt(max(rs*cs,1e-12))
      float s = wave_sum(term);
      if (lane == 0) atomicAdd(&srow[t * 64 + r], s);
    }
  }
  __syncthreads();
  for (int i = tid; i < kT * 64; i += 256) {
    int t = i >> 6, r = i & 63;
    atomicAdd(&S[(size_t)t * kN + r0 + r], srow[i]);
  }
}

// -------- aggregation GEMM: V = M_pos@y_pos - M_neg@y_neg, loss += V^2
// grid (kN/16); block 256 = 8 waves; block = 16 rows x 256 cols, wave = 2 col
// tiles. A fragments (shared by all 8 waves) built cooperatively in LDS in
// WMMA fragment layout: exp/normalize once per element per block.
__global__ void k_agg(const float* __restrict__ distP, const float* __restrict__ distN,
                      const half_t* __restrict__ ypT, const half_t* __restrict__ ynT,
                      const float* __restrict__ meansum, const float* __restrict__ wrs,
                      const float* __restrict__ wcsn, const float* __restrict__ wcsp,
                      const float* __restrict__ Sneg, const float* __restrict__ Spos,
                      float* out) {
  const float Ts[kT] = {0.05f, 0.1f, 0.2f};
  int tid = threadIdx.x, wave = tid >> 5, lane = tid & 31;
  int i0 = blockIdx.x * 16;
  int n0 = wave * 32;
  float mean = meansum[0] * (1.f / ((float)kN * (float)kN));
  float ninv[kT];
  #pragma unroll
  for (int t = 0; t < kT; t++) ninv[t] = -1.f / (Ts[t] * mean);

  __shared__ float sDp[16 * 32], sDn[16 * 32];          // dist tiles
  __shared__ float sWp[kT * 32], sWn[kT * 32];          // col weights
  __shared__ float sCP[kT * 16], sKP[kT * 16];          // per-row coefs (pos)
  __shared__ float sCN[kT * 16], sKN[kT * 16];          // per-row coefs (neg)
  __shared__ __align__(32) half_t sAp[512];             // A frag (pos), lane*16+h
  __shared__ __align__(32) half_t sAn[512];             // A frag (neg)

  // per-row coefficients for rows i0..i0+15 (once per block)
  if (tid < kT * 16) {
    int t = tid >> 4, m = tid & 15;
    float w  = wrs[(size_t)t * kN + i0 + m];
    float sn = Sneg[(size_t)t * kN + i0 + m];
    float sp = Spos[(size_t)t * kN + i0 + m];
    sCP[tid] = sn * w;  sKP[tid] = sn * 1e6f;  // min(cP*wcs, kP) = sn*min(wrs*wcs,1e6)
    sCN[tid] = sp * w;  sKN[tid] = sp * 1e6f;
  }

  const half_t* bp0 = ypT + (size_t)(n0 + (lane & 15)) * kN + ((lane & 16) ? 16 : 0);
  const half_t* bp1 = bp0 + (size_t)16 * kN;
  const half_t* bn0 = ynT + (size_t)(n0 + (lane & 15)) * kN + ((lane & 16) ? 16 : 0);
  const half_t* bn1 = bn0 + (size_t)16 * kN;

  // cooperative A-build mapping: positions p=2*tid, 2*tid+1 of fragment
  int pL = (tid * 2) >> 4;        // fragment lane 0..31
  int pm = pL & 15;               // matrix row within tile
  int ph = (tid * 2) & 15;        // even half index

  v8f aPa = {}, aPb = {}, aNa = {}, aNb = {};
  for (int k0 = 0; k0 < kN; k0 += 32) {
    __syncthreads();                                   // protect prior reads
    #pragma unroll
    for (int q = 0; q < 2; q++) {                      // coalesced dist tile stage
      int id = tid + q * 256;
      int rr = id >> 5, cc = id & 31;
      sDp[id] = distP[(size_t)(i0 + rr) * kN + k0 + cc];
      sDn[id] = distN[(size_t)(i0 + rr) * kN + k0 + cc];
    }
    if (tid < kT * 32) {
      sWp[tid] = wcsp[(size_t)(tid >> 5) * kN + k0 + (tid & 31)];
      sWn[tid] = wcsn[(size_t)(tid >> 5) * kN + k0 + (tid & 31)];
    }
    __syncthreads();

    // build 2 adjacent fragment halves per matrix per thread
    v2h vp, vn;
    #pragma unroll
    for (int q = 0; q < 2; q++) {
      int h = ph + q;
      int k = h + ((pL & 16) ? 16 : 0);  // contiguous K map (matches B)
      float dp = sDp[pm * 32 + k];
      float dn = sDn[pm * 32 + k];
      float mp = 0.f, mn = 0.f;
      #pragma unroll
      for (int t = 0; t < kT; t++) {
        float ep = __expf(fmaxf(dp * ninv[t], -80.f));
        mp += fminf(sCP[t * 16 + pm] * sWp[t * 32 + k], sKP[t * 16 + pm]) * ep;
        float en = __expf(fmaxf(dn * ninv[t], -80.f));
        mn += fminf(sCN[t * 16 + pm] * sWn[t * 32 + k], sKN[t * 16 + pm]) * en;
      }
      vp[q] = (half_t)mp;
      vn[q] = (half_t)mn;
    }
    *(v2h*)(sAp + tid * 2) = vp;                       // single b32 LDS store
    *(v2h*)(sAn + tid * 2) = vn;
    __syncthreads();

    v16h Av_p = *(const v16h*)(sAp + lane * 16);       // 2x ds_load_b128
    v16h Av_n = *(const v16h*)(sAn + lane * 16);
    v16h Bpa = *(const v16h*)(bp0 + k0);
    v16h Bpb = *(const v16h*)(bp1 + k0);
    v16h Bna = *(const v16h*)(bn0 + k0);
    v16h Bnb = *(const v16h*)(bn1 + k0);
    aPa = __builtin_amdgcn_wmma_f32_16x16x32_f16(false, Av_p, false, Bpa, (short)0, aPa, false, false);
    aPb = __builtin_amdgcn_wmma_f32_16x16x32_f16(false, Av_p, false, Bpb, (short)0, aPb, false, false);
    aNa = __builtin_amdgcn_wmma_f32_16x16x32_f16(false, Av_n, false, Bna, (short)0, aNa, false, false);
    aNb = __builtin_amdgcn_wmma_f32_16x16x32_f16(false, Av_n, false, Bnb, (short)0, aNb, false, false);
  }

  float lsum = 0.f;
  #pragma unroll
  for (int r = 0; r < 8; r++) {
    float v = aPa[r] - aNa[r]; lsum += v * v;
    v = aPb[r] - aNb[r];       lsum += v * v;
  }
  float s = wave_sum(lsum);
  if (lane == 0) atomicAdd(out, s * (1.f / ((float)kN * (float)kD)));
}

// ---------------------------------------------------------------------------
extern "C" void kernel_launch(void* const* d_in, const int* in_sizes, int n_in,
                              void* d_out, int out_size, void* d_ws, size_t ws_size,
                              hipStream_t stream) {
  (void)in_sizes; (void)n_in; (void)out_size; (void)ws_size;
  const float* x  = (const float*)d_in[0];
  const float* yp = (const float*)d_in[1];
  const float* yn = (const float*)d_in[2];
  float* out = (float*)d_out;
  float* ws  = (float*)d_ws;

  float* meansum = ws;                         // [16] (pad)
  float* rs   = ws + 16;                       // [kT*kN]
  float* csn  = rs   + (size_t)kT * kN;
  float* csp  = csn  + (size_t)kT * kN;
  float* Sneg = csp  + (size_t)kT * kN;
  float* Spos = Sneg + (size_t)kT * kN;
  int accFloats = 16 + 5 * kT * kN;            // region zeroed each launch
  float* wrs  = ws + accFloats;                // rsqrt(rs|csn|csp), contiguous
  float* wcsn = wrs  + (size_t)kT * kN;
  float* wcsp = wcsn + (size_t)kT * kN;
  float* nx   = wcsp + (size_t)kT * kN;
  float* nyp  = nx  + kN;
  float* nyn  = nyp + kN;
  size_t off = (size_t)(nyn + kN - ws);
  off = (off + 63) & ~(size_t)63;              // 256B-align dist matrices
  float* distN_ = ws + off;                    // [kN*kN]
  float* distP_ = distN_ + (size_t)kN * kN;    // [kN*kN]
  half_t* xh  = (half_t*)(distP_ + (size_t)kN * kN);
  half_t* yph = xh  + (size_t)kN * kD;
  half_t* ynh = yph + (size_t)kN * kD;
  half_t* ypT = ynh + (size_t)kN * kD;         // y_pos^T (kD x kN) fp16
  half_t* ynT = ypT + (size_t)kN * kD;         // y_neg^T

  k_zero <<<64, 256, 0, stream>>>(ws, accFloats, out);
  k_prep <<<3 * kN, 256, 0, stream>>>(x, yp, yn, xh, yph, ynh, ypT, ynT, nx, nyp, nyn);
  k_dist <<<dim3(kN / 128, kN / 16, 2), 256, 0, stream>>>(xh, yph, ynh, nx, nyp, nyn,
                                                          distP_, distN_, meansum);
  k_sums <<<dim3(kN / 256, kN / 64, 2), 256, 0, stream>>>(distP_, distN_, meansum,
                                                          rs, csn, csp);
  k_rsqrt<<<(3 * kT * kN + 255) / 256, 256, 0, stream>>>(rs, wrs, 3 * kT * kN);
  k_wsums<<<dim3(kN / 256, kN / 64, 2), 256, 0, stream>>>(distP_, distN_, meansum,
                                                          wrs, wcsn, wcsp, Sneg, Spos);
  k_agg  <<<kN / 16, 256, 0, stream>>>(distP_, distN_, ypT, ynT, meansum,
                                       wrs, wcsn, wcsp, Sneg, Spos, out);
}
